// PrimeKnetFullStream_12567074308651
// MI455X (gfx1250) — matrix operand order repaired and checked by reading, not compiled, not run on hardware
//
#include <hip/hip_runtime.h>
#include <math.h>
#include <stdint.h>

// ---------------------------------------------------------------------------
// PrimeKnet full-stream for MI455X (gfx1250, wave32).
// DFT / iDFT are GEMMs executed with v_wmma_f32_16x16x32_f16 (f32 accumulate).
// A operand is streamed into LDS by the Tensor Data Mover (tensor_load_to_lds,
// double-buffered, s_wait_tensorcnt pipelined); B panel staged once in LDS in
// WMMA fragment layout (ds_load_b128 per fragment). Everything else is fused
// memory-bound elementwise kernels.
// ---------------------------------------------------------------------------

#define BATCH   8
#define LIN     320000
#define TWIN    400          // WIN == NFFT
#define HOPS    100
#define TFR     3197         // (LIN - NFFT)/HOP + 1
#define FREQ    201          // NFFT/2 + 1
#define NM      (BATCH*TFR)  // 25576 frames
#define MPAD    25584        // round up to 16
#define KP      416          // K padded (400 -> 416, 13 steps of 32)
#define KSTEPS  13
#define NSPEC   402          // [re | im]
#define NPAD    416          // N padded for forward GEMM
#define NIDFT   400          // iDFT output length (multiple of 16)
#define LOUTC   319600       // cropped output length
#define NSEQ    (BATCH*FREQ) // 1608 sequences in LKFCA stage
#define TFQ     (TFR*FREQ)   // 642597 elements per (b,c) map
#define NCHUNK  64
#define CHUNKSZ ((TFQ + NCHUNK - 1)/NCHUNK)

#define TWOPI 6.283185307179586476925f

typedef __attribute__((ext_vector_type(16))) _Float16 v16h;
typedef __attribute__((ext_vector_type(8)))  _Float16 v8h;
typedef __attribute__((ext_vector_type(8)))  float    v8f;
typedef __attribute__((ext_vector_type(4)))  unsigned int v4u;
typedef __attribute__((ext_vector_type(8)))  int      v8i;
typedef __attribute__((ext_vector_type(4)))  int      v4i;

#if defined(__has_builtin)
#if __has_builtin(__builtin_amdgcn_tensor_load_to_lds)
#define USE_TDM 1
#endif
#endif
#ifndef USE_TDM
#define USE_TDM 0
#endif

// --------------------------- packed param layout ---------------------------
#define P_DEC_B 0
#define P_DEC_W 2
#define P_ENC(e) (6 + 46*(e))     // cb(2) cw(36) pb(2) prelu(2) pw(4)
#define P_LK(l)  (98 + 230*(l))
// within LK block:
//  dw_b @+0(4)  dw_w @+4(12)
//  branches (k=3,5,7,11) @ +16,+38,+68,+106 : adb(2) adw(2k) apb(2) apw(4) cdb(2) cdw(2k)
//  g.nb @+160  g.ng @+162  pf_b @+164(8)  pf_w @+172(16)  pl_b @+188  pl_w @+190(16) scale @+206
//  nb @+208  ng @+210  pw1_b @+212(4)  pw1_w @+216(8)  pw2_b @+224  pw2_w @+226(4)

// ------------------------------ ws layout ----------------------------------
static constexpr size_t AL(size_t x){ return (x + 255) & ~(size_t)255; }
static constexpr size_t OFF_P     = 0;
static constexpr size_t SZ_P      = AL(558*4);
static constexpr size_t OFF_STATS = OFF_P + SZ_P;   // mean(16) istd(16) partials(16*64 float2)
static constexpr size_t SZ_STATS  = AL(128 + (size_t)16*NCHUNK*8);
static constexpr size_t OFF_A16   = OFF_STATS + SZ_STATS;             // f16 GEMM A (frames / spectrum)
static constexpr size_t SZ_A16    = AL((size_t)MPAD*KP*2);
static constexpr size_t OFF_BF    = OFF_A16 + SZ_A16;                 // f16 forward DFT basis KPxNPAD
static constexpr size_t SZ_BF     = AL((size_t)KP*NPAD*2);
static constexpr size_t OFF_BI    = OFF_BF + SZ_BF;                   // f16 inverse DFT basis KPxNIDFT
static constexpr size_t SZ_BI     = AL((size_t)KP*NIDFT*2);
static constexpr size_t OFF_C32   = OFF_BI + SZ_BI;                   // f32 GEMM C (spectrum / frames)
static constexpr size_t SZ_C32    = AL((size_t)MPAD*NPAD*4);
static constexpr size_t OFF_X0    = OFF_C32 + SZ_C32;                 // (B,2,T,F)
static constexpr size_t SZ_X      = AL((size_t)BATCH*2*TFQ*4);
static constexpr size_t OFF_X1    = OFF_X0 + SZ_X;
static constexpr size_t OFF_HA    = OFF_X1 + SZ_X;                    // (B*F,2,T)
static constexpr size_t OFF_HY    = OFF_HA + SZ_X;
static constexpr size_t OFF_HB    = OFF_HY + SZ_X;

// --------------------------------------------------------------------------
__device__ __forceinline__ float hannw(int n){
  return 0.5f*(1.0f - cosf(TWOPI * (float)n / (float)TWIN));
}
__device__ __forceinline__ float sigm(float x){ return 1.0f/(1.0f + expf(-x)); }

__device__ __forceinline__ void ln2(float y0, float y1,
                                    float ng0, float ng1, float nb0, float nb1,
                                    float& n0, float& n1){
  float m  = 0.5f*(y0 + y1);
  float d0 = y0 - m, d1 = y1 - m;
  float var = 0.5f*(d0*d0 + d1*d1);
  float is  = rsqrtf(var + 1e-5f);
  n0 = d0*is*ng0 + nb0;
  n1 = d1*is*ng1 + nb1;
}

// ------------------------- basis generation (f16) --------------------------
__global__ void init_basis_kernel(_Float16* __restrict__ BF, _Float16* __restrict__ BI){
  int idx = blockIdx.x*256 + threadIdx.x;
  const int nF = KP*NPAD;
  if (idx < nF){
    int k = idx / NPAD, n = idx % NPAD;   // k = time sample, n = output bin col
    float v = 0.f;
    if (k < TWIN && n < NSPEC){
      int j  = (n < FREQ) ? n : (n - FREQ);
      int ph = (k * j) % TWIN;            // exact integer phase reduction
      float ang = TWOPI * (float)ph / (float)TWIN;
      v = (n < FREQ) ? cosf(ang) : -sinf(ang);   // rfft: Re = sum x cos, Im = -sum x sin
    }
    BF[idx] = (_Float16)v;
  } else {
    int id2 = idx - nF;
    if (id2 < KP*NIDFT){
      int k = id2 / NIDFT, n = id2 % NIDFT; // k = spectral row, n = time col
      float v = 0.f;
      if (k < NSPEC){
        int j = (k < FREQ) ? k : (k - FREQ);
        float c = (j == 0 || j == 200) ? 1.f : 2.f;
        int ph = (j * n) % TWIN;
        float ang = TWOPI * (float)ph / (float)TWIN;
        v = ((k < FREQ) ? c*cosf(ang) : -c*sinf(ang)) / (float)TWIN;
      }
      BI[id2] = (_Float16)v;
    }
  }
}

// ---------------------- framing + Hann window -> f16 A ---------------------
__global__ void frame_window_kernel(const float* __restrict__ audio, _Float16* __restrict__ A){
  int idx = blockIdx.x*256 + threadIdx.x;
  if (idx >= MPAD*KP) return;
  int m = idx / KP, k = idx % KP;
  float v = 0.f;
  if (m < NM && k < TWIN){
    int b = m / TFR, t = m % TFR;
    v = audio[(size_t)b*LIN + t*HOPS + k] * hannw(k);
  }
  A[idx] = (_Float16)v;
}

#if USE_TDM
// Issue one TDM 2D tile load: 128 rows x 32 cols of f16 from global (row
// stride = lda elements) into LDS at byte offset ldsByte (rows compacted to
// 64B). rowsAvail < 128 makes the TDM zero-fill the tail (OOB reads -> 0).
// D# layout per CDNA5 ISA 8.3/8.4 (group0: count/lds/global/type=2;
// group1: data_size=2B, dims/strides; groups 2-3 unused for 2D).
// This toolchain exposes the 6-arg builtin form:
//   (uint32x4 g0, int32x8 g1, int32x4 g2, int32x4 g3, int32x8 g4, i32 cpol)
__device__ __forceinline__ void tdm_load_a(const _Float16* gsrc, unsigned ldsByte,
                                           int rowsAvail, int lda){
  unsigned long long ga = (unsigned long long)(uintptr_t)gsrc;
  v4u g0;
  g0[0] = 1u;                                            // count=1 (valid), user load
  g0[1] = ldsByte;                                       // lds_addr
  g0[2] = (unsigned)(ga & 0xffffffffu);                  // global_addr[31:0]
  g0[3] = (unsigned)((ga >> 32) & 0x1ffffffu) | (2u<<30);// global_addr[56:32] | type=2
  v8i g1;
  g1[0] = (1 << 16);                                     // data_size=1 (2 bytes)
  g1[1] = (32 << 16);                                    // tensor_dim0[15:0]=32
  g1[2] = ((rowsAvail & 0xffff) << 16);                  // tensor_dim0 hi | tensor_dim1 lo
  g1[3] = (32 << 16);                                    // tensor_dim1 hi | tile_dim0=32
  g1[4] = 128;                                           // tile_dim1=128, tile_dim2=0
  g1[5] = lda;                                           // tensor_dim0_stride[31:0]
  g1[6] = 0;
  g1[7] = 0;
  v4i g2 = {0,0,0,0};
  v4i g3 = {0,0,0,0};
  v8i g4 = {0,0,0,0,0,0,0,0};
  __builtin_amdgcn_tensor_load_to_lds(g0, g1, g2, g3, g4, 0);
}
#endif

// --------------------------- WMMA GEMM (f16->f32) --------------------------
// 256 threads = 8 waves; each wave owns one 16x16 C tile (8 consecutive
// M-tiles per workgroup, one N-tile per blockIdx.y). The B panel
// (KP x 16 f16, 13 KB) is staged once into LDS pre-swizzled into WMMA
// B-fragment layout so each lane reads its 16-half fragment with two
// ds_load_b128. A k-slabs (128x32 f16, 8 KB) are streamed into a
// double-buffered LDS stage by the Tensor Data Mover, pipelined with
// s_wait_tensorcnt; fragments are two contiguous 16B runs -> ds_load_b128.
//
// Lane layouts (ISA 7.12.2):
//  A 16x32 f16 : lane L holds row M=L%16; halves 0..7 -> K=(L<16?0:8)+j,
//                halves 8..15 -> K=16+(L<16?0:8)+(j-8)  (two 16B runs).
//  B 32x16 f16 : lane L holds col N=L%16; half j -> same K as A's half j (table order).
//  C/D f32     : lane L holds col N=L%16; VGPR v -> row M=(L<16?0:8)+v.
__global__ void __launch_bounds__(256)
wmma_dft_gemm(const _Float16* __restrict__ A, const _Float16* __restrict__ Bm,
              float* __restrict__ C, int lda, int ldb, int ldc, int Mtiles){
  __shared__ _Float16 sB[KSTEPS*32*16];   // [step][lane][halves 0..15]
#if USE_TDM
  __shared__ _Float16 sA[2][128*32];      // double-buffered A k-slab
#endif
  const int tid  = threadIdx.x;
  const int lane = tid & 31;
  const int wave = tid >> 5;
  const int nt   = blockIdx.y;
  const int mt   = blockIdx.x*8 + wave;   // wave-uniform
  const int row0 = blockIdx.x*128;

#if USE_TDM
  int rowsAvail = MPAD - row0; if (rowsAvail > 128) rowsAvail = 128;
  if (wave == 0)                               // one DMA per workgroup
    tdm_load_a(A + (size_t)row0*lda, (unsigned)(uintptr_t)&sA[0][0], rowsAvail, lda);
#endif

  // ---- stage B panel into fragment layout (coalesced b128 global reads) ---
  for (int c = tid; c < KP*2; c += 256){
    int k  = c >> 1;
    int h8 = (c & 1)*8;                       // first col of this 8-col chunk
    v8h v = *(const v8h*)(Bm + (size_t)k*ldb + nt*16 + h8);
    int step  = k >> 5;
    int jj    = k & 31;
    int j     = (jj & 7) | ((jj >> 4) << 3);
    int lbase = (jj & 8) ? 16 : 0;
#pragma unroll
    for (int e = 0; e < 8; ++e){
      int lanei = lbase + h8 + e;             // lane that owns column h8+e
      sB[(step*32 + lanei)*16 + j] = v[e];
    }
  }

  const int rl = (lane & 15);
  const int hi = (lane >= 16) ? 1 : 0;
#if !USE_TDM
  __syncthreads();
  int rowA = mt*16 + rl; if (rowA > MPAD - 1) rowA = MPAD - 1;
  const _Float16* abase = A + (size_t)rowA*lda + (hi ? 8 : 0);
#endif
  v8f acc = {};
#pragma unroll
  for (int ks = 0; ks < KSTEPS; ++ks){
    const int kk = ks*32;
#if USE_TDM
    if (wave == 0){
      if (ks + 1 < KSTEPS){                   // prefetch next slab via TDM
        tdm_load_a(A + (size_t)row0*lda + (ks + 1)*32,
                   (unsigned)(uintptr_t)&sA[(ks + 1) & 1][0], rowsAvail, lda);
        __builtin_amdgcn_s_wait_tensorcnt(1); // slab ks retired, ks+1 in flight
      } else {
        __builtin_amdgcn_s_wait_tensorcnt(0); // last slab retired
      }
    }
    __syncthreads();                          // slab ks ready (and B on ks==0)
    const _Float16* ab = &sA[ks & 1][(wave*16 + rl)*32 + (hi ? 8 : 0)];
    v8h a0 = *(const v8h*)ab;                 // ds_load_b128
    v8h a1 = *(const v8h*)(ab + 16);          // ds_load_b128
#else
    __builtin_prefetch(abase + kk + 64, 0, 0);
    v8h a0 = *(const v8h*)(abase + kk);
    v8h a1 = *(const v8h*)(abase + kk + 16);
#endif
    const v8h* bf = (const v8h*)(sB + ((size_t)ks*32 + lane)*16);
    v8h b0 = bf[0], b1 = bf[1];               // ds_load_b128 x2
    v16h av, bv;
#pragma unroll
    for (int e = 0; e < 8; ++e){
      av[e] = a0[e]; av[8+e] = a1[e];
      bv[e] = b0[e]; bv[8+e] = b1[e];
    }
    acc = __builtin_amdgcn_wmma_f32_16x16x32_f16(false, av, false, bv,
                                                 (short)0, acc, false, false);
#if USE_TDM
    __syncthreads();                          // all waves done with slab ks
#endif
  }
  if (mt < Mtiles){
#pragma unroll
    for (int v = 0; v < 8; ++v){
      int ro = mt*16 + (hi ? 8 : 0) + v;
      C[(size_t)ro*ldc + nt*16 + rl] = acc[v];
    }
  }
}

// --------------------------- spectrum -> mag/phase -------------------------
__global__ void magpha_kernel(const float* __restrict__ S, float* __restrict__ X){
  int idx = blockIdx.x*256 + threadIdx.x;
  if (idx >= NM*FREQ) return;
  int m = idx / FREQ, f = idx % FREQ;
  float re = S[(size_t)m*NPAD + f];
  float im = S[(size_t)m*NPAD + FREQ + f];
  int b = m / TFR, t = m % TFR;
  size_t base = (size_t)(b*2)*TFQ + (size_t)t*FREQ + f;
  X[base]               = sqrtf(re*re + im*im);
  X[base + (size_t)TFQ] = atan2f(im, re);
}

// --------------------------- encoder instance-norm -------------------------
__global__ void enc_stats_partial(const float* __restrict__ X, const float* __restrict__ P,
                                  int preluOff, float2* __restrict__ partials){
  int map = blockIdx.y, chunk = blockIdx.x;
  int start = chunk*CHUNKSZ;
  int end = start + CHUNKSZ; if (end > TFQ) end = TFQ;
  float a = P[preluOff + (map & 1)];
  const float* xp = X + (size_t)map*TFQ;
  float s = 0.f, ss = 0.f;
  for (int i = start + threadIdx.x; i < end; i += 256){
    float v = xp[i]; v = (v >= 0.f) ? v : a*v;
    s += v; ss += v*v;
  }
  __shared__ float sh[256], sh2[256];
  sh[threadIdx.x] = s; sh2[threadIdx.x] = ss; __syncthreads();
  for (int o = 128; o > 0; o >>= 1){
    if (threadIdx.x < o){ sh[threadIdx.x] += sh[threadIdx.x+o]; sh2[threadIdx.x] += sh2[threadIdx.x+o]; }
    __syncthreads();
  }
  if (threadIdx.x == 0) partials[map*NCHUNK + chunk] = make_float2(sh[0], sh2[0]);
}

__global__ void enc_stats_final(const float2* __restrict__ partials,
                                float* __restrict__ mean, float* __restrict__ istd){
  int map = threadIdx.x;
  if (map >= 16) return;
  float s = 0.f, ss = 0.f;
  for (int i = 0; i < NCHUNK; ++i){ float2 p = partials[map*NCHUNK + i]; s += p.x; ss += p.y; }
  const float n = (float)TFQ;
  float m = s/n, var = ss/n - m*m;
  mean[map] = m;
  istd[map] = rsqrtf(var + 1e-5f);
}

// fused: residual + pointwise( dilated 3x3 conv ( instnorm( prelu(x) ) ) )
__global__ void enc_conv_kernel(const float* __restrict__ Xin, float* __restrict__ Xout,
                                const float* __restrict__ P, int eb, int dil,
                                const float* __restrict__ mean, const float* __restrict__ istd){
  int idx = blockIdx.x*256 + threadIdx.x;
  if (idx >= NM*FREQ) return;
  int f = idx % FREQ;
  int t = (idx / FREQ) % TFR;
  int b = idx / (FREQ*TFR);
  float cv[2];
#pragma unroll
  for (int co = 0; co < 2; ++co){
    float acc = P[eb + co];                          // cb
#pragma unroll
    for (int ci = 0; ci < 2; ++ci){
      int mp = b*2 + ci;
      float mn = mean[mp], is = istd[mp];
      float al = P[eb + 40 + ci];                    // prelu alpha
#pragma unroll
      for (int kt = 0; kt < 3; ++kt){
        int tt = t + (kt - 2)*dil;
#pragma unroll
        for (int kf = 0; kf < 3; ++kf){
          int ff = f + kf - 1;
          float h = 0.f;
          if (tt >= 0 && ff >= 0 && ff < FREQ){
            float raw = Xin[(size_t)mp*TFQ + (size_t)tt*FREQ + ff];
            float pr  = (raw >= 0.f) ? raw : al*raw;
            h = (pr - mn)*is;
          }
          acc += P[eb + 2 + co*18 + ci*9 + kt*3 + kf] * h;
        }
      }
    }
    cv[co] = acc;
  }
#pragma unroll
  for (int co = 0; co < 2; ++co){
    float h = P[eb + 38 + co] + P[eb + 42 + co*2]*cv[0] + P[eb + 42 + co*2 + 1]*cv[1];
    size_t o = (size_t)(b*2 + co)*TFQ + (size_t)t*FREQ + f;
    Xout[o] = Xin[o] + h;
  }
}

// -------------------- (B,2,T,F) -> (B*F, 2, T) transpose -------------------
__global__ void transpose_to_seq(const float* __restrict__ X, float* __restrict__ H){
  int idx = blockIdx.x*256 + threadIdx.x;
  if (idx >= BATCH*2*TFR*FREQ) return;
  int f = idx % FREQ;
  int t = (idx / FREQ) % TFR;
  int c = (idx / (FREQ*TFR)) % 2;
  int b = idx / (FREQ*TFR*2);
  H[((size_t)(b*FREQ + f)*2 + c)*TFR + t] = X[idx];
}

// ---------------- LKFCA part 1: pw2(GLU(dw(pw1(LN(x))))) + x ---------------
__global__ void lkfca_l1_kernel(const float* __restrict__ Xin, float* __restrict__ Y,
                                const float* __restrict__ P, int lb){
  int idx = blockIdx.x*256 + threadIdx.x;
  if (idx >= NSEQ*TFR) return;
  int n = idx / TFR, t = idx % TFR;
  size_t base = (size_t)n*2*TFR;
  float ng0 = P[lb+210], ng1 = P[lb+211], nb0 = P[lb+208], nb1 = P[lb+209];
  float u[3][4];
#pragma unroll
  for (int s = 0; s < 3; ++s){
    int tau = t - 2 + s;
    if (tau >= 0){
      float x0 = Xin[base + tau], x1 = Xin[base + TFR + tau];
      float n0, n1; ln2(x0, x1, ng0, ng1, nb0, nb1, n0, n1);
#pragma unroll
      for (int o = 0; o < 4; ++o)
        u[s][o] = P[lb+212+o] + P[lb+216+o*2]*n0 + P[lb+216+o*2+1]*n1;
    } else {
#pragma unroll
      for (int o = 0; o < 4; ++o) u[s][o] = 0.f;    // causal zero pad of pw1 output
    }
  }
  float w[4];
#pragma unroll
  for (int c = 0; c < 4; ++c){
    float a = P[lb + c];                            // dw_b
#pragma unroll
    for (int s = 0; s < 3; ++s) a += P[lb + 4 + c*3 + s]*u[s][c];
    w[c] = a;
  }
  float h0 = w[0]*sigm(w[2]);
  float h1 = w[1]*sigm(w[3]);
#pragma unroll
  for (int co = 0; co < 2; ++co){
    float o = P[lb+224+co] + P[lb+226+co*2]*h0 + P[lb+226+co*2+1]*h1;
    Y[base + (size_t)co*TFR + t] = o + Xin[base + (size_t)co*TFR + t];
  }
}

// ---------------- LKFCA part 2 (fused GCGFN): out = (pl+y)*scale + y -------
__global__ void lkfca_l2_kernel(const float* __restrict__ Y, float* __restrict__ Out,
                                const float* __restrict__ P, int lb){
  int idx = blockIdx.x*256 + threadIdx.x;
  if (idx >= NSEQ*TFR) return;
  int n = idx / TFR, t = idx % TFR;
  size_t base = (size_t)n*2*TFR;
  float gng0 = P[lb+162], gng1 = P[lb+163], gnb0 = P[lb+160], gnb1 = P[lb+161];
  const int kk4[4]  = {3, 5, 7, 11};
  const int bro4[4] = {16, 38, 68, 106};
  float outq[8];
#pragma unroll
  for (int i = 0; i < 4; ++i){
    const int k  = kk4[i];
    const int bo = lb + bro4[i];
    const int ch0 = 2*i;
    float ar0 = P[bo],            ar1 = P[bo+1];           // adb
    float vr0 = P[bo + 8 + 2*k],  vr1 = P[bo + 9 + 2*k];   // cdb
    for (int j = 0; j < k; ++j){
      int tau = t - (k - 1) + j;
      if (tau >= 0){
        float y0 = Y[base + tau], y1 = Y[base + TFR + tau];
        float n0, n1; ln2(y0, y1, gng0, gng1, gnb0, gnb1, n0, n1);
        float c0 = P[lb+164+ch0]   + P[lb+172+ch0*2]*n0     + P[lb+172+ch0*2+1]*n1;
        float c1 = P[lb+164+ch0+1] + P[lb+172+(ch0+1)*2]*n0 + P[lb+172+(ch0+1)*2+1]*n1;
        ar0 += P[bo + 2 + j]*c0;
        ar1 += P[bo + 2 + k + j]*c1;
        vr0 += P[bo + 10 + 2*k + j]*c0;
        vr1 += P[bo + 10 + 2*k + k + j]*c1;
      }
    }
    float a0 = P[bo + 2 + 2*k]     + P[bo + 4 + 2*k]*ar0     + P[bo + 4 + 2*k + 1]*ar1;
    float a1 = P[bo + 2 + 2*k + 1] + P[bo + 4 + 2*k + 2]*ar0 + P[bo + 4 + 2*k + 3]*ar1;
    outq[ch0]     = a0*vr0;
    outq[ch0 + 1] = a1*vr1;
  }
#pragma unroll
  for (int co = 0; co < 2; ++co){
    float z = P[lb + 188 + co];
#pragma unroll
    for (int q = 0; q < 8; ++q) z += P[lb + 190 + co*8 + q]*outq[q];
    float yv = Y[base + (size_t)co*TFR + t];
    Out[base + (size_t)co*TFR + t] = (z + yv)*P[lb + 206 + co] + yv;
  }
}

// ------------- decoder 1x1 + polar->cartesian -> f16 GEMM A ---------------
__global__ void decoder_spec_kernel(const float* __restrict__ H, const float* __restrict__ P,
                                    _Float16* __restrict__ A){
  int idx = blockIdx.x*256 + threadIdx.x;
  if (idx >= MPAD*KP) return;
  int m = idx / KP, col = idx % KP;
  float val = 0.f;
  if (m < NM && col < NSPEC){
    int b = m / TFR, t = m % TFR;
    int f = (col < FREQ) ? col : (col - FREQ);
    const float* hp = H + (size_t)(b*FREQ + f)*2*TFR;
    float x0 = hp[t], x1 = hp[TFR + t];
    float em = P[P_DEC_B+0] + P[P_DEC_W+0]*x0 + P[P_DEC_W+1]*x1;
    float ep = P[P_DEC_B+1] + P[P_DEC_W+2]*x0 + P[P_DEC_W+3]*x1;
    val = (col < FREQ) ? em*cosf(ep) : em*sinf(ep);
  }
  A[idx] = (_Float16)val;
}

// ----------------------- windowed OLA + env norm + crop --------------------
__global__ void ola_kernel(const float* __restrict__ FR, float* __restrict__ out){
  int idx = blockIdx.x*256 + threadIdx.x;
  if (idx >= BATCH*LOUTC) return;
  int b = idx / LOUTC, lo = idx % LOUTC;
  int l = lo + TWIN/2;                    // crop offset
  int tmin = (l >= TWIN - 1) ? (l - (TWIN - 1) + HOPS - 1)/HOPS : 0;
  int tmax = l / HOPS; if (tmax > TFR - 1) tmax = TFR - 1;
  float acc = 0.f, env = 0.f;
  for (int t = tmin; t <= tmax; ++t){
    int nn = l - t*HOPS;                  // 0..399
    float w = hannw(nn);
    acc += FR[(size_t)(b*TFR + t)*NPAD + nn] * w;   // irfft * win
    env += w*w;
  }
  out[idx] = acc / fmaxf(env, 1e-11f);
}

// ===========================================================================
extern "C" void kernel_launch(void* const* d_in, const int* in_sizes, int n_in,
                              void* d_out, int out_size, void* d_ws, size_t ws_size,
                              hipStream_t stream) {
  (void)in_sizes; (void)out_size; (void)ws_size;
  char* ws = (char*)d_ws;
  float*     P     = (float*)(ws + OFF_P);
  float*     mean  = (float*)(ws + OFF_STATS);
  float*     istd  = mean + 16;
  float2*    parts = (float2*)(ws + OFF_STATS + 128);
  _Float16*  A16   = (_Float16*)(ws + OFF_A16);
  _Float16*  BF    = (_Float16*)(ws + OFF_BF);
  _Float16*  BI    = (_Float16*)(ws + OFF_BI);
  float*     C32   = (float*)(ws + OFF_C32);
  float*     X0    = (float*)(ws + OFF_X0);
  float*     X1    = (float*)(ws + OFF_X1);
  float*     HA    = (float*)(ws + OFF_HA);
  float*     HY    = (float*)(ws + OFF_HY);
  float*     HB    = (float*)(ws + OFF_HB);
  const float* audio = (const float*)d_in[0];
  float* out = (float*)d_out;

  // --- gather params into packed buffer (JAX pytree order: dict keys sorted)
  static const int kParamSizes[90] = {
    2,4,                                  // dec_b, dec_w
    2,36,2,2,4,                           // enc0: cb,cw,pb,prelu,pw
    2,36,2,2,4,                           // enc1
    4,12, 2,6,2,4,2,6, 2,10,2,4,2,10, 2,14,2,4,2,14, 2,22,2,4,2,22,
    2,2,8,16,2,16,2, 2,2,4,8,2,4,         // lkfca block 0
    4,12, 2,6,2,4,2,6, 2,10,2,4,2,10, 2,14,2,4,2,14, 2,22,2,4,2,22,
    2,2,8,16,2,16,2, 2,2,4,8,2,4          // lkfca block 1
  };
  int nParams = n_in - 1; if (nParams > 90) nParams = 90;
  size_t poff = 0;
  for (int i = 0; i < nParams; ++i){
    (void)hipMemcpyAsync(P + poff, d_in[1 + i], (size_t)kParamSizes[i]*sizeof(float),
                         hipMemcpyDeviceToDevice, stream);
    poff += kParamSizes[i];
  }

  auto G = [](long n){ return (int)((n + 255)/256); };

  // --- STFT: frame+window, forward-DFT GEMM (WMMA), mag/phase
  init_basis_kernel<<<G((long)KP*NPAD + (long)KP*NIDFT), 256, 0, stream>>>(BF, BI);
  frame_window_kernel<<<G((long)MPAD*KP), 256, 0, stream>>>(audio, A16);
  wmma_dft_gemm<<<dim3((MPAD/16 + 7)/8, NPAD/16), 256, 0, stream>>>(A16, BF, C32, KP, NPAD, NPAD, MPAD/16);
  magpha_kernel<<<G((long)NM*FREQ), 256, 0, stream>>>(C32, X0);

  // --- encoder blocks (dil = 1, 2)
  enc_stats_partial<<<dim3(NCHUNK, 16), 256, 0, stream>>>(X0, P, P_ENC(0) + 40, parts);
  enc_stats_final<<<1, 32, 0, stream>>>(parts, mean, istd);
  enc_conv_kernel<<<G((long)NM*FREQ), 256, 0, stream>>>(X0, X1, P, P_ENC(0), 1, mean, istd);
  enc_stats_partial<<<dim3(NCHUNK, 16), 256, 0, stream>>>(X1, P, P_ENC(1) + 40, parts);
  enc_stats_final<<<1, 32, 0, stream>>>(parts, mean, istd);
  enc_conv_kernel<<<G((long)NM*FREQ), 256, 0, stream>>>(X1, X0, P, P_ENC(1), 2, mean, istd);

  // --- LKFCA stage over (B*F, 2, T)
  transpose_to_seq<<<G((long)BATCH*2*TFR*FREQ), 256, 0, stream>>>(X0, HA);
  lkfca_l1_kernel<<<G((long)NSEQ*TFR), 256, 0, stream>>>(HA, HY, P, P_LK(0));
  lkfca_l2_kernel<<<G((long)NSEQ*TFR), 256, 0, stream>>>(HY, HB, P, P_LK(0));
  lkfca_l1_kernel<<<G((long)NSEQ*TFR), 256, 0, stream>>>(HB, HY, P, P_LK(1));
  lkfca_l2_kernel<<<G((long)NSEQ*TFR), 256, 0, stream>>>(HY, HA, P, P_LK(1));

  // --- decoder + ISTFT: polar->cart (f16), inverse-DFT GEMM (WMMA), OLA
  decoder_spec_kernel<<<G((long)MPAD*KP), 256, 0, stream>>>(HA, P, A16);
  wmma_dft_gemm<<<dim3((MPAD/16 + 7)/8, NIDFT/16), 256, 0, stream>>>(A16, BI, C32, KP, NIDFT, NPAD, MPAD/16);
  ola_kernel<<<G((long)BATCH*LOUTC), 256, 0, stream>>>(C32, out);
}
